// TimeDistributed_1159641170154
// MI455X (gfx1250) — matrix operand order repaired and checked by reading, not compiled, not run on hardware
//
#include <hip/hip_runtime.h>
#include <hip/hip_bf16.h>

typedef __attribute__((ext_vector_type(16))) __bf16    bf16x16;
typedef __attribute__((ext_vector_type(8)))  float     floatx8;
typedef __attribute__((ext_vector_type(4)))  unsigned  uint32x4;

union Frag {
    bf16x16  v;
    uint32x4 q[2];
    unsigned u[8];
};

static __device__ __forceinline__ unsigned short f2bf(float f) {
    union { float f; unsigned u; } x; x.f = f;
    unsigned r = x.u + 0x7FFFu + ((x.u >> 16) & 1u);   // round-to-nearest-even
    return (unsigned short)(r >> 16);
}

static __device__ __forceinline__ floatx8 wmma_bf16(const Frag& a, const Frag& b, floatx8 c) {
    return __builtin_amdgcn_wmma_f32_16x16x32_bf16(false, a.v, false, b.v,
                                                   (short)0, c, false, false);
}

// ---------------------------------------------------------------------------
// Pack a row-major f32 weight [K][N] into WMMA B-fragment order (bf16),
// LANE-MAJOR within the fragment: dst[frag*256 + lane*8 + v].
// lane: n = nt*16 + (lane&15); k = kt*32 + (v>>2)*16 + (lane>>4)*8 + 2*(v&3) (+0/+1 packed).
// Each lane's fragment is 32 contiguous bytes -> 2 x b128 loads in the GEMMs.
// ---------------------------------------------------------------------------
__global__ void pack_weight(const float* __restrict__ W, unsigned* __restrict__ dst,
                            int K, int N) {
    int id = blockIdx.x * 256 + threadIdx.x;
    int total = (K / 32) * (N / 16) * 256;
    if (id >= total) return;
    int v    = id & 7;
    int lane = (id >> 3) & 31;
    int frag = id >> 8;
    int ntiles = N / 16;
    int kt = frag / ntiles, nt = frag % ntiles;
    int n = nt * 16 + (lane & 15);
    int k = kt * 32 + ((v >> 2) * 16) + ((lane >> 4) * 8) + (v & 3) * 2;
    unsigned lo = f2bf(W[(size_t)k * N + n]);
    unsigned hi = f2bf(W[(size_t)(k + 1) * N + n]);
    dst[id] = lo | (hi << 16);
}

__global__ void cvt_f32_bf16(const float* __restrict__ src, unsigned short* __restrict__ dst,
                             int n) {
    int i = blockIdx.x * 256 + threadIdx.x;
    if (i < n) dst[i] = f2bf(src[i]);
}

// ---- A-fragment loaders (16x32 bf16, row-major source): 2 x 128-bit loads ----
static __device__ __forceinline__ void loadA_global(Frag& a, const unsigned short* base,
                                                    int lda, int row0, int kt,
                                                    int mA, int koff8) {
    const unsigned short* p = base + (size_t)(row0 + mA) * lda + kt * 32 + koff8;
    a.q[0] = *(const uint32x4*)(p);
    a.q[1] = *(const uint32x4*)(p + 16);
}

static __device__ __forceinline__ void loadA_lds(Frag& a, const unsigned short* base,
                                                 int kt, int mA, int koff8) {
    const unsigned short* p = base + mA * 256 + kt * 32 + koff8;
    a.q[0] = *(const uint32x4*)(p);
    a.q[1] = *(const uint32x4*)(p + 16);
}

// ---- B-fragment loader (pre-swizzled, lane-major): 2 x 128-bit loads ----
static __device__ __forceinline__ void loadB(Frag& b, const unsigned* frags,
                                             int fragIdx, int lane) {
    const unsigned* p = frags + (size_t)fragIdx * 256 + lane * 8;
    b.q[0] = *(const uint32x4*)(p);
    b.q[1] = *(const uint32x4*)(p + 4);
}

// ---------------------------------------------------------------------------
// Gx = x @ [Wz|Wr|Wh] + [bz|br|bh]   (M=65536, K=128, N=768)  one tile / wave
// ---------------------------------------------------------------------------
__global__ void gemm_gx(const unsigned short* __restrict__ xb,
                        const unsigned* __restrict__ wz, const unsigned* __restrict__ wr,
                        const unsigned* __restrict__ wh,
                        const float* __restrict__ bz, const float* __restrict__ br,
                        const float* __restrict__ bh,
                        float* __restrict__ gx) {
    int wid  = (blockIdx.x * 256 + threadIdx.x) >> 5;
    int lane = threadIdx.x & 31;
    int mt = wid / 48;
    int nt = wid % 48;
    int gate = nt >> 4, ntl = nt & 15;
    const unsigned* Bp  = (gate == 0) ? wz : (gate == 1) ? wr : wh;
    const float*    bia = (gate == 0) ? bz : (gate == 1) ? br : bh;

    int mA = lane & 15, koff8 = (lane >> 4) * 8;
    int row0 = mt * 16;
    floatx8 acc = {};
#pragma unroll
    for (int kt = 0; kt < 4; ++kt) {
        Frag a, b;
        loadA_global(a, xb, 128, row0, kt, mA, koff8);
        loadB(b, Bp, kt * 16 + ntl, lane);
        acc = wmma_bf16(a, b, acc);
    }
    int nD = lane & 15, mhi = (lane >> 4) * 8;
    float bv = bia[ntl * 16 + nD];
#pragma unroll
    for (int v = 0; v < 8; ++v) {
        size_t r = (size_t)(row0 + v + mhi);
        gx[r * 768 + gate * 256 + ntl * 16 + nD] = acc[v] + bv;
    }
}

// ---------------------------------------------------------------------------
// ys = hs @ Wo + bo   (M=65536, K=256, N=256)
// ---------------------------------------------------------------------------
__global__ void gemm_y(const unsigned short* __restrict__ hsb,
                       const unsigned* __restrict__ wo, const float* __restrict__ bo,
                       float* __restrict__ ys) {
    int wid  = (blockIdx.x * 256 + threadIdx.x) >> 5;
    int lane = threadIdx.x & 31;
    int mt = wid >> 4;
    int nt = wid & 15;
    int mA = lane & 15, koff8 = (lane >> 4) * 8;
    int row0 = mt * 16;
    floatx8 acc = {};
#pragma unroll
    for (int kt = 0; kt < 8; ++kt) {
        Frag a, b;
        loadA_global(a, hsb, 256, row0, kt, mA, koff8);
        loadB(b, wo, kt * 16 + nt, lane);
        acc = wmma_bf16(a, b, acc);
    }
    int nD = lane & 15, mhi = (lane >> 4) * 8;
    float bv = bo[nt * 16 + nD];
#pragma unroll
    for (int v = 0; v < 8; ++v) {
        size_t r = (size_t)(row0 + v + mhi);
        ys[r * 256 + nt * 16 + nD] = acc[v] + bv;
    }
}

// ---------------------------------------------------------------------------
// Recurrent scan: 2 persistent blocks x 256 threads; block = 16 batch rows.
// h kept in LDS (f32 + bf16); U matrices streamed from L2 (pre-swizzled bf16).
// ---------------------------------------------------------------------------
__global__ void __launch_bounds__(256, 1)
gru_scan(const float* __restrict__ h0, const float* __restrict__ gx,
         const unsigned* __restrict__ uz, const unsigned* __restrict__ ur,
         const unsigned* __restrict__ uh,
         float* __restrict__ hs_out, unsigned short* __restrict__ hs_bf) {
    __shared__ float          hF[16 * 256];
    __shared__ unsigned short hB[16 * 256];
    __shared__ unsigned short rhB[16 * 256];

    int b0 = blockIdx.x * 16;
    for (int i = threadIdx.x; i < 16 * 256; i += 256) {
        float v = h0[(size_t)(b0 + (i >> 8)) * 256 + (i & 255)];
        hF[i] = v;
        hB[i] = f2bf(v);
    }
    __syncthreads();

    int lane = threadIdx.x & 31;
    int w    = threadIdx.x >> 5;
    int mA = lane & 15, koff8 = (lane >> 4) * 8;   // A-fragment addressing
    int nD = lane & 15, mhi   = (lane >> 4) * 8;   // C/D-fragment addressing

    for (int t = 0; t < 2048; ++t) {
        floatx8 zz[2];
        // ---- phase 1: z, r, stage r*h ----
#pragma unroll
        for (int tau = 0; tau < 2; ++tau) {
            int nt = w * 2 + tau;
            floatx8 rr;
#pragma unroll
            for (int g = 0; g < 2; ++g) {
                const unsigned* U = g ? ur : uz;
                floatx8 acc = {};
#pragma unroll
                for (int kt = 0; kt < 8; ++kt) {
                    Frag a, b;
                    loadA_lds(a, hB, kt, mA, koff8);
                    loadB(b, U, kt * 16 + nt, lane);
                    acc = wmma_bf16(a, b, acc);
                }
                floatx8 s;
#pragma unroll
                for (int v = 0; v < 8; ++v) {
                    size_t row = (size_t)(b0 + v + mhi) * 2048 + t;
                    float pre = acc[v] + gx[row * 768 + g * 256 + nt * 16 + nD];
                    s[v] = 1.0f / (1.0f + __expf(-pre));
                }
                if (g == 0) zz[tau] = s; else rr = s;
            }
#pragma unroll
            for (int v = 0; v < 8; ++v) {
                int m = v + mhi, n = nt * 16 + nD;
                rhB[m * 256 + n] = f2bf(rr[v] * hF[m * 256 + n]);
            }
        }
        __syncthreads();
        // ---- phase 2: hh = tanh(gxh + (r*h)@Uh), hn, writeback ----
#pragma unroll
        for (int tau = 0; tau < 2; ++tau) {
            int nt = w * 2 + tau;
            floatx8 acc = {};
#pragma unroll
            for (int kt = 0; kt < 8; ++kt) {
                Frag a, b;
                loadA_lds(a, rhB, kt, mA, koff8);
                loadB(b, uh, kt * 16 + nt, lane);
                acc = wmma_bf16(a, b, acc);
            }
#pragma unroll
            for (int v = 0; v < 8; ++v) {
                int m = v + mhi, n = nt * 16 + nD;
                size_t row = (size_t)(b0 + m) * 2048 + t;
                float hh = tanhf(acc[v] + gx[row * 768 + 512 + nt * 16 + nD]);
                float zv = zz[tau][v];
                float hn = (1.0f - zv) * hF[m * 256 + n] + zv * hh;
                hs_out[row * 256 + n] = hn;
                hs_bf[row * 256 + n]  = f2bf(hn);
                hF[m * 256 + n] = hn;
                hB[m * 256 + n] = f2bf(hn);
            }
        }
        __syncthreads();
    }
}

// ---------------------------------------------------------------------------
extern "C" void kernel_launch(void* const* d_in, const int* in_sizes, int n_in,
                              void* d_out, int out_size, void* d_ws, size_t ws_size,
                              hipStream_t stream) {
    (void)in_sizes; (void)n_in; (void)out_size; (void)ws_size;
    const float* x  = (const float*)d_in[0];
    const float* h0 = (const float*)d_in[1];
    const float* Wz = (const float*)d_in[2];
    const float* Uz = (const float*)d_in[3];
    const float* bz = (const float*)d_in[4];
    const float* Wr = (const float*)d_in[5];
    const float* Ur = (const float*)d_in[6];
    const float* br = (const float*)d_in[7];
    const float* Wh = (const float*)d_in[8];
    const float* Uh = (const float*)d_in[9];
    const float* bh = (const float*)d_in[10];
    const float* Wo = (const float*)d_in[11];
    const float* bo = (const float*)d_in[12];

    const size_t BT = 32ull * 2048ull;             // 65536 rows
    const size_t nWfrag = 4ull * 16ull * 256ull;   // [128,256] swizzled dwords
    const size_t nUfrag = 8ull * 16ull * 256ull;   // [256,256] swizzled dwords

    unsigned* wsWz = (unsigned*)d_ws;
    unsigned* wsWr = wsWz + nWfrag;
    unsigned* wsWh = wsWr + nWfrag;
    unsigned* wsUz = wsWh + nWfrag;
    unsigned* wsUr = wsUz + nUfrag;
    unsigned* wsUh = wsUr + nUfrag;
    unsigned* wsWo = wsUh + nUfrag;
    unsigned short* xb  = (unsigned short*)(wsWo + nUfrag);
    float*          gx  = (float*)(xb + BT * 128);
    unsigned short* hsb = (unsigned short*)(gx + BT * 768);

    float* ys = (float*)d_out;
    float* hs = ys + BT * 256;

    // 1) pre-swizzle weights into WMMA B-fragment bf16 layout (lane-major)
    pack_weight<<<64,  256, 0, stream>>>(Wz, wsWz, 128, 256);
    pack_weight<<<64,  256, 0, stream>>>(Wr, wsWr, 128, 256);
    pack_weight<<<64,  256, 0, stream>>>(Wh, wsWh, 128, 256);
    pack_weight<<<128, 256, 0, stream>>>(Uz, wsUz, 256, 256);
    pack_weight<<<128, 256, 0, stream>>>(Ur, wsUr, 256, 256);
    pack_weight<<<128, 256, 0, stream>>>(Uh, wsUh, 256, 256);
    pack_weight<<<128, 256, 0, stream>>>(Wo, wsWo, 256, 256);

    // 2) x -> bf16
    int nx = (int)(BT * 128);
    cvt_f32_bf16<<<(nx + 255) / 256, 256, 0, stream>>>(x, xb, nx);

    // 3) batched input-gate GEMM: Gx = x @ [Wz|Wr|Wh] + biases
    gemm_gx<<<24576, 256, 0, stream>>>(xb, wsWz, wsWr, wsWh, bz, br, bh, gx);

    // 4) sequential GRU scan (2 persistent workgroups, 16 batch rows each)
    gru_scan<<<2, 256, 0, stream>>>(h0, gx, wsUz, wsUr, wsUh, hs, hsb);

    // 5) batched output projection: ys = hs @ Wo + bo
    gemm_y<<<8192, 256, 0, stream>>>(hsb, wsWo, bo, ys);
}
